// PositionHead_45896020525333
// MI455X (gfx1250) — compile-verified, matrix operand order checked by reading, not run on hardware
//
#include <hip/hip_runtime.h>
#include <hip/hip_bf16.h>
#include <cstdint>
#include <cstddef>

#define EDIM 768
#define NHEAD 12
#define HD 64
#define BSZ 8
#define NN 512

typedef __attribute__((ext_vector_type(16))) __bf16 v16bf;
typedef __attribute__((ext_vector_type(8)))  float  v8f;

union FragBF {
    v16bf v;
    uint4 u[2];
};

// Async global->LDS copy of 16 bytes (CDNA5 GLOBAL_LOAD_ASYNC_TO_LDS_B128).
// ldsoff = workgroup-relative LDS byte offset (low 32 bits of generic ptr).
__device__ __forceinline__ void async_copy_b128(unsigned ldsoff, const void* gptr) {
    asm volatile("global_load_async_to_lds_b128 %0, %1, off"
                 :: "v"(ldsoff), "v"(gptr) : "memory");
}
__device__ __forceinline__ void async_wait0() {
    asm volatile("s_wait_asynccnt 0x0" ::: "memory");
}
__device__ __forceinline__ unsigned lds_off(const void* p) {
    return (unsigned)(size_t)p;     // AS(3) offset lives in the low 32 bits
}

// ---------------------------------------------------------------------------
// Kernel 1a: x[b,i,e] = (bf16) query[i,b,e]   (swapaxes + downconvert)
// ---------------------------------------------------------------------------
__global__ __launch_bounds__(256)
void prep_x_kernel(const float* __restrict__ query, __bf16* __restrict__ xb) {
    int idx = blockIdx.x * 256 + threadIdx.x;          // over BSZ*NN*EDIM
    if (idx >= BSZ * NN * EDIM) return;
    int e  = idx % EDIM;
    int bn = idx / EDIM;
    int i  = bn % NN;
    int b  = bn / NN;
    xb[idx] = (__bf16)query[((size_t)i * BSZ + b) * EDIM + e];
}

// ---------------------------------------------------------------------------
// Kernel 1b: WT[j,k] = (bf16) W[k,j] for Wq/Wk/Wv
// ---------------------------------------------------------------------------
__global__ __launch_bounds__(256)
void prep_w_kernel(const float* __restrict__ Wq, const float* __restrict__ Wk,
                   const float* __restrict__ Wv,
                   __bf16* __restrict__ wqT, __bf16* __restrict__ wkT,
                   __bf16* __restrict__ wvT) {
    int idx = blockIdx.x * 256 + threadIdx.x;          // over EDIM*EDIM
    if (idx >= EDIM * EDIM) return;
    int k = idx % EDIM;
    int j = idx / EDIM;
    size_t src = (size_t)k * EDIM + j;
    wqT[idx] = (__bf16)Wq[src];
    wkT[idx] = (__bf16)Wk[src];
    wvT[idx] = (__bf16)Wv[src];
}

// ---------------------------------------------------------------------------
// Kernel 2: out(M=BSZ*NN, 768) = (A(M,768) @ WT^T + bias) * scale, bf16 out.
// ---------------------------------------------------------------------------
__global__ __launch_bounds__(256)
void gemm_qkv_kernel(const __bf16* __restrict__ A,    // (M, 768) row-major bf16
                     const __bf16* __restrict__ BT,   // (768, 768) row-major (W^T)
                     const float*  __restrict__ bias, // (768)
                     float scale,
                     __bf16* __restrict__ out)        // (M, 768) bf16
{
    int wave = threadIdx.x >> 5;
    int lane = threadIdx.x & 31;
    int lm   = lane & 15;
    int lhi  = lane >> 4;                               // K-half select

    int m0 = blockIdx.x * 64 + (wave & 3) * 16;
    int n0 = blockIdx.y * 32 + (wave >> 2) * 16;

    const __bf16* arow = A  + (size_t)(m0 + lm) * EDIM; // row M=lm of tile
    const __bf16* brow = BT + (size_t)(n0 + lm) * EDIM; // column N=lm of tile

    v8f acc = {};
    #pragma unroll
    for (int k0 = 0; k0 < EDIM; k0 += 32) {
        FragBF a, bb;
        a.u[0]  = *(const uint4*)(arow + k0 +      lhi * 8);
        a.u[1]  = *(const uint4*)(arow + k0 + 16 + lhi * 8);
        bb.u[0] = *(const uint4*)(brow + k0 +      lhi * 8);
        bb.u[1] = *(const uint4*)(brow + k0 + 16 + lhi * 8);
        acc = __builtin_amdgcn_wmma_f32_16x16x32_bf16(
                  false, a.v, false, bb.v, (short)0, acc, false, false);
    }

    int ncol = n0 + lm;
    float bv = bias[ncol];
    #pragma unroll
    for (int r = 0; r < 8; ++r) {
        int mm = m0 + r + lhi * 8;
        float val = (acc[r] + bv) * scale;
        out[(size_t)mm * EDIM + ncol] = (__bf16)val;
    }
}

// ---------------------------------------------------------------------------
// Kernel 3: t[c,b,h,j] = sum_d v[b,j,h*64+d] * Wc[h*64+d]
// ---------------------------------------------------------------------------
__global__ __launch_bounds__(256)
void t_kernel(const __bf16* __restrict__ vb,
              const float* __restrict__ W1, const float* __restrict__ W2,
              const float* __restrict__ W3, float* __restrict__ t)
{
    int idx = blockIdx.x * 256 + threadIdx.x;           // over 3*BSZ*NHEAD*NN
    if (idx >= 3 * BSZ * NHEAD * NN) return;
    int j = idx % NN;  int rest = idx / NN;
    int h = rest % NHEAD; rest /= NHEAD;
    int b = rest % BSZ;
    int c = rest / BSZ;
    const float* W = (c == 0) ? W1 : (c == 1 ? W2 : W3);
    const __bf16* vrow = vb + ((size_t)(b * NN + j)) * EDIM + h * HD;
    const float*  w    = W + h * HD;
    float s = 0.f;
    #pragma unroll
    for (int d = 0; d < HD; ++d) s += (float)vrow[d] * w[d];
    t[idx] = s;
}

// ---------------------------------------------------------------------------
// Kernel 4: fused attention + rotational aggregation + output projection.
// Grid: (NN/16, BSZ). Block: 256 threads (8 waves), 16-row i-tile.
// ---------------------------------------------------------------------------
__global__ __launch_bounds__(256)
void attn_kernel(const __bf16* __restrict__ qb,   // (b,n,E) bf16, pre-scaled
                 const __bf16* __restrict__ kb,   // (b,n,E) bf16
                 const float*  __restrict__ bias, // (b,H,n,n)
                 const float*  __restrict__ pos,  // (b,n,3)
                 const float*  __restrict__ t,    // (3,b,H,n)
                 const float*  __restrict__ b1, const float* __restrict__ b2,
                 const float*  __restrict__ b3,
                 float* __restrict__ out)         // (b,n,3)
{
    __shared__ __bf16 sh_q[16][EDIM];        // 24 KB  (all heads of the i-tile)
    __shared__ __bf16 sh_k[NN][HD];          // 64 KB  (one head's K)
    __shared__ float  sh_s[16][NN];          // 32 KB  (score tile)
    __shared__ float  sh_delta[3][16][NN];   // 96 KB  (unit direction vectors)
    __shared__ float  sh_t[3][NN];           //  6 KB  (current head's t vectors)
    __shared__ float  sh_red[16][16];
    __shared__ float  sh_rowmax[16];
    __shared__ float  sh_rowsum[16];
    __shared__ float  sh_fred[3][16][16];    //  3 KB

    const int b   = blockIdx.y;
    const int i0  = blockIdx.x * 16;
    const int tid = threadIdx.x;
    const int wave = tid >> 5, lane = tid & 31;
    const int lm = lane & 15, lhi = lane >> 4;
    const int r = tid & 15, g = tid >> 4;            // softmax row / j-chunk

    // ---- async stage q tile (contiguous 16*768 bf16 = 1536 x b128) ----
    {
        const char* src = (const char*)(qb + ((size_t)(b * NN + i0)) * EDIM);
        unsigned dst = lds_off(&sh_q[0][0]);
        #pragma unroll
        for (int p = tid; p < 16 * EDIM / 8; p += 256)
            async_copy_b128(dst + p * 16, src + p * 16);
    }

    // ---- precompute unit delta vectors (head-invariant) ----
    {
        float pix = pos[((size_t)b * NN + i0 + r) * 3 + 0];
        float piy = pos[((size_t)b * NN + i0 + r) * 3 + 1];
        float piz = pos[((size_t)b * NN + i0 + r) * 3 + 2];
        #pragma unroll 4
        for (int jj = 0; jj < 32; ++jj) {
            int j = g * 32 + jj;
            float px = pos[((size_t)b * NN + j) * 3 + 0];
            float py = pos[((size_t)b * NN + j) * 3 + 1];
            float pz = pos[((size_t)b * NN + j) * 3 + 2];
            float dx = px - pix, dy = py - piy, dz = pz - piz;
            float dist = sqrtf(dx * dx + dy * dy + dz * dz);
            float rinv = 1.0f / (dist + 1e-5f);
            sh_delta[0][r][j] = dx * rinv;
            sh_delta[1][r][j] = dy * rinv;
            sh_delta[2][r][j] = dz * rinv;
        }
    }
    async_wait0();
    __syncthreads();

    float facc0 = 0.f, facc1 = 0.f, facc2 = 0.f;

    for (int h = 0; h < NHEAD; ++h) {
        // ---- async stage K for this head: 512 rows x 64 bf16 (128 B/row) ----
        {
            const __bf16* ksrc = kb + ((size_t)b * NN) * EDIM + h * HD;
            unsigned dst = lds_off(&sh_k[0][0]);
            #pragma unroll
            for (int p = tid; p < NN * HD / 8; p += 256) {   // p: b128 index
                int j = p >> 3, q4 = p & 7;
                async_copy_b128(dst + p * 16,
                                ksrc + (size_t)j * EDIM + q4 * 8);
            }
        }
        // ---- stage this head's t vectors (overlaps the async copy) ----
        for (int p = tid; p < 3 * NN; p += 256) {
            int c = p / NN, j = p % NN;
            sh_t[c][j] = t[(((size_t)c * BSZ + b) * NHEAD + h) * NN + j];
        }
        async_wait0();
        __syncthreads();

        // ---- WMMA: each wave computes 4 j-tiles of 16 ----
        const float* brow = bias + ((size_t)(b * NHEAD + h)) * NN * NN;
        for (int jt = wave * 4; jt < wave * 4 + 4; ++jt) {
            v8f acc = {};
            const __bf16* qrow = &sh_q[lm][h * HD];
            const __bf16* krow = &sh_k[jt * 16 + lm][0];
            #pragma unroll
            for (int k0 = 0; k0 < HD; k0 += 32) {
                FragBF a, bb;
                a.u[0]  = *(const uint4*)(qrow + k0 +      lhi * 8);
                a.u[1]  = *(const uint4*)(qrow + k0 + 16 + lhi * 8);
                bb.u[0] = *(const uint4*)(krow + k0 +      lhi * 8);
                bb.u[1] = *(const uint4*)(krow + k0 + 16 + lhi * 8);
                acc = __builtin_amdgcn_wmma_f32_16x16x32_bf16(
                          false, a.v, false, bb.v, (short)0, acc, false, false);
            }
            int jcol = jt * 16 + lm;
            #pragma unroll
            for (int r2 = 0; r2 < 8; ++r2) {
                int mm = r2 + lhi * 8;
                sh_s[mm][jcol] = acc[r2] + brow[(size_t)(i0 + mm) * NN + jcol];
            }
        }
        __syncthreads();

        // ---- softmax over j=512 (rows split into 16 chunks of 32) ----
        float lmax = -3.4e38f;
        #pragma unroll
        for (int jj = 0; jj < 32; ++jj)
            lmax = fmaxf(lmax, sh_s[r][g * 32 + jj]);
        sh_red[r][g] = lmax;
        __syncthreads();
        if (g == 0) {
            float mx = sh_red[r][0];
            #pragma unroll
            for (int p = 1; p < 16; ++p) mx = fmaxf(mx, sh_red[r][p]);
            sh_rowmax[r] = mx;
        }
        __syncthreads();
        float mx = sh_rowmax[r];
        float pe[32];
        float lsum = 0.f;
        #pragma unroll
        for (int jj = 0; jj < 32; ++jj) {
            float e = __expf(sh_s[r][g * 32 + jj] - mx);
            pe[jj] = e;
            lsum += e;
        }
        sh_red[r][g] = lsum;
        __syncthreads();
        if (g == 0) {
            float s = 0.f;
            #pragma unroll
            for (int p = 0; p < 16; ++p) s += sh_red[r][p];
            sh_rowsum[r] = s;
        }
        __syncthreads();
        float inv = 1.0f / sh_rowsum[r];

        // ---- delta/t contraction for this head (LDS-resident operands) ----
        float a0 = 0.f, a1 = 0.f, a2 = 0.f;
        #pragma unroll
        for (int jj = 0; jj < 32; ++jj) {
            int j = g * 32 + jj;
            float p = pe[jj];
            a0 += p * sh_delta[0][r][j] * sh_t[0][j];
            a1 += p * sh_delta[1][r][j] * sh_t[1][j];
            a2 += p * sh_delta[2][r][j] * sh_t[2][j];
        }
        facc0 += a0 * inv;
        facc1 += a1 * inv;
        facc2 += a2 * inv;
        __syncthreads();   // protect sh_k / sh_s / sh_t before next head
    }

    // ---- final reduce over 16 j-chunks, add output bias, store ----
    sh_fred[0][r][g] = facc0;
    sh_fred[1][r][g] = facc1;
    sh_fred[2][r][g] = facc2;
    __syncthreads();
    if (tid < 48) {
        int c = tid / 16, r2 = tid % 16;
        float s = 0.f;
        #pragma unroll
        for (int p = 0; p < 16; ++p) s += sh_fred[c][r2][p];
        float bc = (c == 0) ? b1[0] : (c == 1 ? b2[0] : b3[0]);
        out[((size_t)b * NN + i0 + r2) * 3 + c] = s + bc;
    }
}

// ---------------------------------------------------------------------------
extern "C" void kernel_launch(void* const* d_in, const int* in_sizes, int n_in,
                              void* d_out, int out_size, void* d_ws, size_t ws_size,
                              hipStream_t stream) {
    const float* query    = (const float*)d_in[0];
    const float* att_bias = (const float*)d_in[1];
    const float* node_pos = (const float*)d_in[2];
    const float* Wq = (const float*)d_in[3];
    const float* bq = (const float*)d_in[4];
    const float* Wk = (const float*)d_in[5];
    const float* bk = (const float*)d_in[6];
    const float* Wv = (const float*)d_in[7];
    const float* bv = (const float*)d_in[8];
    const float* W1 = (const float*)d_in[9];
    const float* b1 = (const float*)d_in[10];
    const float* W2 = (const float*)d_in[11];
    const float* b2 = (const float*)d_in[12];
    const float* W3 = (const float*)d_in[13];
    const float* b3 = (const float*)d_in[14];
    float* out = (float*)d_out;

    // workspace carve-out (all offsets 256B aligned)
    char* ws = (char*)d_ws;
    __bf16* xb  = (__bf16*)ws; ws += (size_t)BSZ * NN * EDIM * 2;      // 6 MB
    __bf16* wqT = (__bf16*)ws; ws += (size_t)EDIM * EDIM * 2;
    __bf16* wkT = (__bf16*)ws; ws += (size_t)EDIM * EDIM * 2;
    __bf16* wvT = (__bf16*)ws; ws += (size_t)EDIM * EDIM * 2;
    __bf16* qb  = (__bf16*)ws; ws += (size_t)BSZ * NN * EDIM * 2;
    __bf16* kb  = (__bf16*)ws; ws += (size_t)BSZ * NN * EDIM * 2;
    __bf16* vb  = (__bf16*)ws; ws += (size_t)BSZ * NN * EDIM * 2;
    float*  tb  = (float*)ws;  ws += (size_t)3 * BSZ * NHEAD * NN * 4;

    const float scaling = 0.125f;   // HD^-0.5 = 64^-0.5

    prep_x_kernel<<<(BSZ * NN * EDIM + 255) / 256, 256, 0, stream>>>(query, xb);
    prep_w_kernel<<<(EDIM * EDIM + 255) / 256, 256, 0, stream>>>(
        Wq, Wk, Wv, wqT, wkT, wvT);

    dim3 ggrid(BSZ * NN / 64, EDIM / 32);
    gemm_qkv_kernel<<<ggrid, 256, 0, stream>>>(xb, wqT, bq, scaling, qb);
    gemm_qkv_kernel<<<ggrid, 256, 0, stream>>>(xb, wkT, bk, 1.0f, kb);
    gemm_qkv_kernel<<<ggrid, 256, 0, stream>>>(xb, wvT, bv, 1.0f, vb);

    t_kernel<<<(3 * BSZ * NHEAD * NN + 255) / 256, 256, 0, stream>>>(
        vb, W1, W2, W3, tb);

    dim3 agrid(NN / 16, BSZ);
    attn_kernel<<<agrid, 256, 0, stream>>>(qb, kb, att_bias, node_pos, tb,
                                           b1, b2, b3, out);
}
